// Attention_22368189678055
// MI455X (gfx1250) — matrix operand order repaired and checked by reading, not compiled, not run on hardware
//
#include <hip/hip_runtime.h>

typedef __attribute__((ext_vector_type(16))) __bf16 v16bf;
typedef __attribute__((ext_vector_type(8)))  float  v8f;
typedef unsigned int u32x4 __attribute__((ext_vector_type(4)));
typedef int          i32x8 __attribute__((ext_vector_type(8)));
typedef int          i32x4 __attribute__((ext_vector_type(4)));

#if __has_builtin(__builtin_amdgcn_tensor_load_to_lds) && \
    __has_builtin(__builtin_amdgcn_s_wait_tensorcnt)
#define HAVE_TDM 1
#endif

union BPack { unsigned int u[8]; v16bf v; };

__device__ inline unsigned short f2bf(float f) {
  unsigned int u = __builtin_bit_cast(unsigned int, f);
  u += 0x7FFFu + ((u >> 16) & 1u);
  return (unsigned short)(u >> 16);
}

__device__ inline v8f vzero8() {
  v8f z;
#pragma unroll
  for (int i = 0; i < 8; ++i) z[i] = 0.f;
  return z;
}

__device__ inline v8f wmma_bf16(const BPack& a, const BPack& b, v8f c) {
  return __builtin_amdgcn_wmma_f32_16x16x32_bf16(
      false, a.v, false, b.v, (short)0, c, false, false);
}

// A operand 16x32 (16-bit): lane r = row; VGPR j holds K pair at
// kk = 2*(j&3) + 16*(j>>2) + 8*half   (ISA 7.12.2, 16-bit A layout)
__device__ inline BPack load_a(const unsigned short* tile, int ld, int lane) {
  int r = lane & 15, h = lane >> 4;
  const unsigned short* row = tile + r * ld;
  BPack x;
#pragma unroll
  for (int j = 0; j < 8; ++j) {
    int kk = 2 * (j & 3) + 16 * (j >> 2) + 8 * h;
    x.u[j] = *(const unsigned int*)(row + kk);
  }
  return x;
}

// B operand 32x16 (16-bit): lane n = col; VGPR j holds K pair at
// kk = 2*(j&3) + 16*(j>>2) + 8*half (same K order as A)
__device__ inline BPack load_b(const unsigned short* tile, int ld, int lane) {
  int n = lane & 15, h = lane >> 4;
  const unsigned short* row = tile + n * ld;
  BPack x;
#pragma unroll
  for (int j = 0; j < 8; ++j)
    x.u[j] = *(const unsigned int*)(row + 2 * (j & 3) + 16 * (j >> 2) + 8 * h);
  return x;
}

// ---------------------------------------------------------------------------
// Bias precompute: biasp[h][qr][kc] over padded 208x208 grid, -1e30 in pads
// ---------------------------------------------------------------------------
__global__ __launch_bounds__(256) void build_bias(const float* __restrict__ tab,
                                                  const int* __restrict__ idx,
                                                  float* __restrict__ bp) {
  int e = blockIdx.x * 256 + threadIdx.x;
  if (e >= 12 * 208 * 208) return;
  int h = e / (208 * 208);
  int rem = e - h * (208 * 208);
  int r = rem / 208;
  int c = rem - r * 208;
  float v = -1e30f;
  if (r < 197 && c < 197) v = tab[idx[r * 197 + c] * 12 + h];
  bp[e] = v;
}

// ---------------------------------------------------------------------------
// GEMM: out[m][n] = A[m][k] * W[n][k] + bvec[n]
// 128x128 tile, 8 waves as 2(M)x4(N), K-step 32, bf16 WMMA, register
// double-buffering so next K-step's global loads overlap the WMMAs.
// mode 0: qkv epilogue -> scatter bf16 q(scaled)/k/v  [b][h][tok][d]
// mode 1: proj epilogue -> f32 out
// ---------------------------------------------------------------------------
__global__ __launch_bounds__(256) void gemm_wmma(
    const float* __restrict__ Af, const unsigned short* __restrict__ Ab,
    const float* __restrict__ W, const float* __restrict__ bvec,
    float* __restrict__ outf,
    unsigned short* __restrict__ qo, unsigned short* __restrict__ ko,
    unsigned short* __restrict__ vo, int M, int N, int K, int mode) {
  __shared__ unsigned short As[128 * 40];
  __shared__ unsigned short Bs[128 * 40];

  const int tid = threadIdx.x;
  const int tm = blockIdx.y * 128, tn = blockIdx.x * 128;
  const int wid = tid >> 5, lane = tid & 31;
  const int wm = wid >> 2, wn = wid & 3;

  v8f acc[4][2];
#pragma unroll
  for (int i = 0; i < 4; ++i) {
    acc[i][0] = vzero8();
    acc[i][1] = vzero8();
  }

  const int sr = tid >> 1;        // staging row 0..127
  const int cs = (tid & 1) * 16;  // staging col segment

  const unsigned short* abase = Ab ? Ab + (size_t)(tm + sr) * K + cs : (const unsigned short*)0;
  const float* afbase = Af ? Af + (size_t)(tm + sr) * K + cs : (const float*)0;
  const float* wbase = W + (size_t)(tn + sr) * K + cs;

  float fa[16], fb[16];
  uint4 ua[2];

  // prefetch K-step 0
  if (Ab) {
    ua[0] = *(const uint4*)(abase);
    ua[1] = *(const uint4*)(abase + 8);
  } else {
#pragma unroll
    for (int q = 0; q < 16; ++q) fa[q] = afbase[q];
  }
#pragma unroll
  for (int q = 0; q < 16; ++q) fb[q] = wbase[q];

  for (int k0 = 0; k0 < K; k0 += 32) {
    // ---- commit staged data to LDS as bf16 ----
    uint4* da = (uint4*)(&As[sr * 40 + cs]);
    if (Ab) {
      da[0] = ua[0];
      da[1] = ua[1];
    } else {
      unsigned int p[8];
#pragma unroll
      for (int q = 0; q < 8; ++q)
        p[q] = (unsigned int)f2bf(fa[2 * q]) |
               ((unsigned int)f2bf(fa[2 * q + 1]) << 16);
      da[0] = make_uint4(p[0], p[1], p[2], p[3]);
      da[1] = make_uint4(p[4], p[5], p[6], p[7]);
    }
    {
      unsigned int p[8];
#pragma unroll
      for (int q = 0; q < 8; ++q)
        p[q] = (unsigned int)f2bf(fb[2 * q]) |
               ((unsigned int)f2bf(fb[2 * q + 1]) << 16);
      uint4* db = (uint4*)(&Bs[sr * 40 + cs]);
      db[0] = make_uint4(p[0], p[1], p[2], p[3]);
      db[1] = make_uint4(p[4], p[5], p[6], p[7]);
    }
    __syncthreads();

    // ---- prefetch next K-step while WMMAs run ----
    const int kn = k0 + 32;
    if (kn < K) {
      if (Ab) {
        ua[0] = *(const uint4*)(abase + kn);
        ua[1] = *(const uint4*)(abase + kn + 8);
      } else {
#pragma unroll
        for (int q = 0; q < 16; ++q) fa[q] = afbase[kn + q];
      }
#pragma unroll
      for (int q = 0; q < 16; ++q) fb[q] = wbase[kn + q];
    }

    BPack bt0 = load_b(&Bs[(wn * 32) * 40], 40, lane);
    BPack bt1 = load_b(&Bs[(wn * 32 + 16) * 40], 40, lane);
#pragma unroll
    for (int i = 0; i < 4; ++i) {
      BPack at = load_a(&As[(wm * 64 + i * 16) * 40], 40, lane);
      acc[i][0] = wmma_bf16(at, bt0, acc[i][0]);
      acc[i][1] = wmma_bf16(at, bt1, acc[i][1]);
    }
    __syncthreads();
  }

  // ---- epilogue ----
  const int half = lane >> 4, col = lane & 15;
#pragma unroll
  for (int i = 0; i < 4; ++i)
#pragma unroll
    for (int j = 0; j < 2; ++j)
#pragma unroll
      for (int rr = 0; rr < 8; ++rr) {
        int m = tm + wm * 64 + i * 16 + rr + 8 * half;
        int n = tn + wn * 32 + j * 16 + col;
        float val = acc[i][j][rr] + bvec[n];
        if (mode == 0) {
          int which = n / 768, rem = n - which * 768;
          int hh = rem >> 6, d = rem & 63;
          int b = m / 197, tok = m - b * 197;
          size_t dst = (((size_t)b * 12 + hh) * 197 + tok) * 64 + d;
          if (which == 0)       qo[dst] = f2bf(val * 0.125f);  // q * hd^-0.5
          else if (which == 1)  ko[dst] = f2bf(val);
          else                  vo[dst] = f2bf(val);
        } else {
          outf[(size_t)m * N + n] = val;
        }
      }
}

// ---------------------------------------------------------------------------
// Flash attention per (b,h). 4 waves, each owns 16-query blocks qb = wid+4k.
// K staged into LDS [208][68] by the Tensor Data Mover (row pad 4 elements
// via TDM pad_interval/pad_amount), V transposed [64][226] (tokens padded to
// 224), per-wave P buffer [16][36] for the C->A layout round trip.
// ---------------------------------------------------------------------------
#define KS_LD 68
#define VT_LD 226
#define PB_LD 36

__global__ __launch_bounds__(128) void attn_wmma(
    const unsigned short* __restrict__ qg, const unsigned short* __restrict__ kg,
    const unsigned short* __restrict__ vg, const float* __restrict__ biasp,
    unsigned short* __restrict__ og) {
  __shared__ unsigned short Ks[208 * KS_LD];
  __shared__ unsigned short Vt[64 * VT_LD];
  __shared__ unsigned short Pb[4 * 16 * PB_LD];

  const int bh = blockIdx.x;
  const int b = bh / 12, h = bh - b * 12;
  const int tid = threadIdx.x;
  const int wid = tid >> 5, lane = tid & 31;
  const int half = lane >> 4, col = lane & 15;

  const size_t base = (size_t)bh * 197 * 64;
  const float* bias_h = biasp + (size_t)h * 208 * 208;

#ifdef HAVE_TDM
  // ---- K tile via Tensor Data Mover: 2D tensor 197x64 (dim0=d, dim1=tok),
  // data_size=2B, LDS pad 2 dwords every 32 dwords -> row stride 68 elems.
  if (wid == 0) {
    unsigned long long ga = (unsigned long long)(const void*)(kg + base);
    unsigned ldsa = (unsigned)(size_t)(&Ks[0]);
    u32x4 g0;
    g0[0] = 1u;                                   // count=1 (valid descriptor)
    g0[1] = ldsa;                                 // lds_addr
    g0[2] = (unsigned)ga;                         // global_addr[31:0]
    g0[3] = (unsigned)((ga >> 32) & 0x1FFFFFFu)   // global_addr[56:32]
            | (2u << 30);                         // type=2 ("image")
    i32x8 g1;
    g1[0] = (int)((1u << 16)      // data_size = 2 bytes
                  | (1u << 20)    // pad_enable
                  | (4u << 22)    // pad_interval: 32 dwords
                  | (1u << 25));  // pad_amount: 2 dwords
    g1[1] = (int)(64u << 16);     // tensor_dim0 = 64
    g1[2] = (int)(197u << 16);    // tensor_dim1 = 197
    g1[3] = (int)(64u << 16);     // tile_dim0 = 64
    g1[4] = 197;                  // tile_dim1 = 197 (tile_dim2 = 0)
    g1[5] = 64;                   // tensor_dim0_stride = 64
    g1[6] = 0;
    g1[7] = 0;
    i32x4 gz4;
    gz4[0] = 0; gz4[1] = 0; gz4[2] = 0; gz4[3] = 0;
    i32x8 gz8;
#pragma unroll
    for (int q = 0; q < 8; ++q) gz8[q] = 0;
    __builtin_amdgcn_tensor_load_to_lds(g0, g1, gz4, gz4, gz8, 0);
  }
  // zero pad rows 197..207 (disjoint from TDM destination region)
  for (int idx = tid; idx < 11 * KS_LD; idx += 128) Ks[197 * KS_LD + idx] = 0;
#else
  // stage K: Ks[tok][d], zero rows 197..207
  for (int idx = tid; idx < 208 * 16; idx += 128) {
    int tok = idx >> 4, seg = (idx & 15) * 4;
    unsigned int w0 = 0, w1 = 0;
    if (tok < 197) {
      const unsigned int* s = (const unsigned int*)(kg + base + (size_t)tok * 64 + seg);
      w0 = s[0]; w1 = s[1];
    }
    unsigned int* d = (unsigned int*)(&Ks[tok * KS_LD + seg]);
    d[0] = w0; d[1] = w1;
  }
#endif

  // stage V transposed: Vt[d][tok], tokens 0..223 (zeros >= 197)
  for (int idx = tid; idx < 224 * 16; idx += 128) {
    int tok = idx >> 4, seg = (idx & 15) * 4;
    unsigned int w0 = 0, w1 = 0;
    if (tok < 197) {
      const unsigned int* s = (const unsigned int*)(vg + base + (size_t)tok * 64 + seg);
      w0 = s[0]; w1 = s[1];
    }
    Vt[(seg + 0) * VT_LD + tok] = (unsigned short)(w0 & 0xFFFFu);
    Vt[(seg + 1) * VT_LD + tok] = (unsigned short)(w0 >> 16);
    Vt[(seg + 2) * VT_LD + tok] = (unsigned short)(w1 & 0xFFFFu);
    Vt[(seg + 3) * VT_LD + tok] = (unsigned short)(w1 >> 16);
  }

#ifdef HAVE_TDM
  __builtin_amdgcn_s_wait_tensorcnt(0);
#endif
  __syncthreads();

  unsigned short* pb = &Pb[wid * 16 * PB_LD];

  for (int qb = wid; qb < 13; qb += 4) {
    const int qbase = qb * 16;

    // Q A-tiles (d 0..31 and 32..63), rows clamped to valid tokens
    BPack qlo, qhi;
    {
      int tok = qbase + (lane & 15);
      if (tok > 196) tok = 196;
      const unsigned short* row = qg + base + (size_t)tok * 64;
#pragma unroll
      for (int j = 0; j < 8; ++j) {
        int kk = 2 * (j & 3) + 16 * (j >> 2) + 8 * half;
        qlo.u[j] = *(const unsigned int*)(row + kk);
        qhi.u[j] = *(const unsigned int*)(row + 32 + kk);
      }
    }

    v8f o[4];
#pragma unroll
    for (int d = 0; d < 4; ++d) o[d] = vzero8();
    float mrow[8], lrow[8];
#pragma unroll
    for (int rr = 0; rr < 8; ++rr) { mrow[rr] = -1e30f; lrow[rr] = 0.f; }

    for (int pair = 0; pair < 7; ++pair) {
      const int kb0 = pair * 2;
      const bool solo = (pair == 6);
      const int kb1 = solo ? kb0 : kb0 + 1;  // clamped: stays in-bounds

      v8f s0 = vzero8(), s1 = vzero8();
      {
        BPack blo = load_b(&Ks[(kb0 * 16) * KS_LD], KS_LD, lane);
        BPack bhi = load_b(&Ks[(kb0 * 16) * KS_LD + 32], KS_LD, lane);
        s0 = wmma_bf16(qlo, blo, s0);
        s0 = wmma_bf16(qhi, bhi, s0);
      }
      {
        BPack blo = load_b(&Ks[(kb1 * 16) * KS_LD], KS_LD, lane);
        BPack bhi = load_b(&Ks[(kb1 * 16) * KS_LD + 32], KS_LD, lane);
        s1 = wmma_bf16(qlo, blo, s1);
        s1 = wmma_bf16(qhi, bhi, s1);
      }

      // online softmax update (row stats reduced over 16-lane halves)
#pragma unroll
      for (int rr = 0; rr < 8; ++rr) {
        int row = qbase + rr + 8 * half;
        float x0 = s0[rr] + bias_h[row * 208 + kb0 * 16 + col];
        float x1 = (solo ? -1e30f : s1[rr]) + bias_h[row * 208 + kb1 * 16 + col];
        float t = fmaxf(x0, x1);
#pragma unroll
        for (int off = 1; off < 16; off <<= 1)
          t = fmaxf(t, __shfl_xor(t, off, 16));
        float mn = fmaxf(mrow[rr], t);
        float alpha = __expf(mrow[rr] - mn);
        mrow[rr] = mn;
        float p0 = __expf(x0 - mn);
        float p1 = __expf(x1 - mn);  // 0 when solo (x1 ~ -1e30)
        float rs = p0 + p1;
#pragma unroll
        for (int off = 1; off < 16; off <<= 1)
          rs += __shfl_xor(rs, off, 16);
        lrow[rr] = lrow[rr] * alpha + rs;
#pragma unroll
        for (int d = 0; d < 4; ++d) o[d][rr] *= alpha;
        int lr = rr + 8 * half;
        pb[lr * PB_LD + col] = f2bf(p0);
        pb[lr * PB_LD + 16 + col] = f2bf(p1);
      }
      asm volatile("s_wait_dscnt 0" ::: "memory");

      // O += P @ V  (B operand from Vt: tokens kb0*16 .. kb0*16+31)
      BPack at = load_a(pb, PB_LD, lane);
#pragma unroll
      for (int d = 0; d < 4; ++d) {
        BPack bt = load_b(&Vt[(d * 16) * VT_LD + kb0 * 16], VT_LD, lane);
        o[d] = wmma_bf16(at, bt, o[d]);
      }
    }

    // normalize and store bf16 to attn_out [b][tok][h*64+d]
#pragma unroll
    for (int rr = 0; rr < 8; ++rr) {
      int row = qbase + rr + 8 * half;
      if (row < 197) {
        float inv = 1.0f / lrow[rr];
#pragma unroll
        for (int d = 0; d < 4; ++d) {
          int dd = d * 16 + col;
          og[((size_t)b * 197 + row) * 768 + h * 64 + dd] = f2bf(o[d][rr] * inv);
        }
      }
    }
  }
}

// ---------------------------------------------------------------------------
extern "C" void kernel_launch(void* const* d_in, const int* in_sizes, int n_in,
                              void* d_out, int out_size, void* d_ws, size_t ws_size,
                              hipStream_t stream) {
  (void)in_sizes; (void)n_in; (void)out_size; (void)ws_size;
  const float* x       = (const float*)d_in[0];
  const float* qkv_w   = (const float*)d_in[1];
  const float* qkv_b   = (const float*)d_in[2];
  const float* proj_w  = (const float*)d_in[3];
  const float* proj_b  = (const float*)d_in[4];
  const float* rel_tab = (const float*)d_in[5];
  const int*   rel_idx = (const int*)d_in[6];
  float* out = (float*)d_out;

  const size_t QKV_ELEMS = (size_t)128 * 12 * 197 * 64;  // per-tensor q/k/v
  char* ws = (char*)d_ws;
  size_t off = 0;
  auto take = [&](size_t bytes) -> void* {
    void* p = ws + off;
    off = (off + bytes + 255) & ~(size_t)255;
    return p;
  };
  unsigned short* qb = (unsigned short*)take(QKV_ELEMS * 2);
  unsigned short* kb = (unsigned short*)take(QKV_ELEMS * 2);
  unsigned short* vb = (unsigned short*)take(QKV_ELEMS * 2);
  unsigned short* ao = (unsigned short*)take((size_t)25216 * 768 * 2);
  float* biasp = (float*)take((size_t)12 * 208 * 208 * 4);

  // 1) padded bias gather
  build_bias<<<dim3((12 * 208 * 208 + 255) / 256), dim3(256), 0, stream>>>(
      rel_tab, rel_idx, biasp);

  // 2) QKV projection: [25216,768] x [2304,768]^T -> q/k/v bf16
  gemm_wmma<<<dim3(18, 197), dim3(256), 0, stream>>>(
      x, nullptr, qkv_w, qkv_b, nullptr, qb, kb, vb, 25216, 2304, 768, 0);

  // 3) flash attention per (b,h)
  attn_wmma<<<dim3(128 * 12), dim3(128), 0, stream>>>(qb, kb, vb, biasp, ao);

  // 4) output projection: [25216,768] x [768,768]^T -> f32 out
  gemm_wmma<<<dim3(6, 197), dim3(256), 0, stream>>>(
      nullptr, ao, proj_w, proj_b, out, nullptr, nullptr, nullptr,
      25216, 768, 768, 1);
}
